// TinyLM_2070174237100
// MI455X (gfx1250) — compile-verified, matrix operand order checked by reading, not run on hardware
//
#include <hip/hip_runtime.h>
#include <math.h>

typedef __bf16        v8bf  __attribute__((ext_vector_type(8)));
typedef __bf16        v16bf __attribute__((ext_vector_type(16)));
typedef float         v8f   __attribute__((ext_vector_type(8)));
typedef unsigned int  u32x4 __attribute__((ext_vector_type(4)));
typedef int           i32x8 __attribute__((ext_vector_type(8)));
typedef int           i32x4 __attribute__((ext_vector_type(4)));

#define BB 4
#define TT 1024
#define CC 256
#define HH 16
#define LL 8
#define VV 32000
#define MM (BB*TT)   // 4096 token rows

// ---------------- helpers ----------------
__device__ __forceinline__ unsigned short f2bf(float f) {
    unsigned int u = __float_as_uint(f);
    u += 0x7FFFu + ((u >> 16) & 1u);          // round-to-nearest-even
    return (unsigned short)(u >> 16);
}
__device__ __forceinline__ unsigned int pk2(float a, float b) {
    return (unsigned int)f2bf(a) | ((unsigned int)f2bf(b) << 16);
}

// Tensor Data Mover: DMA a [128 rows x 32 elem] bf16 tile (rows of a [N][K]
// row-major tensor) from global into LDS at lds_off. Uniform per workgroup;
// issue from one wave only.  D# layout per CDNA5 ISA ch.8 (2D tensor, groups
// 2/3 zero).  This toolchain uses the 6-arg builtin (clang-23 / therock-10.0).
__device__ __forceinline__ void tdm_load_tile_b(const unsigned short* gsrc,
                                                unsigned lds_off, int K, int N) {
    unsigned long long ga = (unsigned long long)(size_t)gsrc;
    u32x4 g0;
    g0[0] = 1u;                                             // count=1 (valid), user mode
    g0[1] = lds_off;                                        // lds_addr (bytes)
    g0[2] = (unsigned)(ga & 0xFFFFFFFFu);                   // global_addr[31:0]
    g0[3] = (unsigned)((ga >> 32) & 0x01FFFFFFu) | (2u << 30); // addr[56:32] | type=2
    i32x8 g1;
    g1[0] = 0x00010000;                                     // data_size=1 (2B), wg_mask=0
    g1[1] = (K & 0xFFFF) << 16;                             // tensor_dim0[15:0]
    g1[2] = ((unsigned)K >> 16) | ((N & 0xFFFF) << 16);     // dim0 hi | tensor_dim1 lo
    g1[3] = ((unsigned)N >> 16) | (32u << 16);              // dim1 hi | tile_dim0=32
    g1[4] = 128;                                            // tile_dim1=128, tile_dim2=0
    g1[5] = K;                                              // tensor_dim0_stride[31:0]
    g1[6] = 0;                                              // stride hi | dim1_stride (2D: unused)
    g1[7] = 0;
    i32x4 z4 = {0, 0, 0, 0};
    i32x8 z8 = {0, 0, 0, 0, 0, 0, 0, 0};
    __builtin_amdgcn_tensor_load_to_lds(g0, g1, z4, z4, z8, 0); // tracked by TENSORcnt
}

// ---------------- weight transpose + fp32->bf16 ----------------
// W: [K][N] fp32 (row-major), WT: [N][K] bf16. grid.z = layer (stride K*N).
__global__ __launch_bounds__(256) void convT_kernel(const float* __restrict__ W,
                                                    unsigned short* __restrict__ WT,
                                                    int K, int N) {
    size_t per = (size_t)K * N;
    const float* Wl = W + per * blockIdx.z;
    unsigned short* WTl = WT + per * blockIdx.z;
    int i = blockIdx.x * 256 + threadIdx.x;
    int n = i / K, k = i % K;
    WTl[(size_t)n * K + k] = f2bf(Wl[(size_t)k * N + n]);
}

// ---------------- embedding ----------------
__global__ __launch_bounds__(256) void embed_kernel(const int* __restrict__ idx,
                                                    const float* __restrict__ tok,
                                                    const float* __restrict__ pos,
                                                    float* __restrict__ x) {
    int i = blockIdx.x * 256 + threadIdx.x;     // over MM*CC
    int row = i >> 8;                            // CC = 256
    int c   = i & 255;
    int t   = row & (TT - 1);
    x[i] = tok[(size_t)idx[row] * CC + c] + pos[(size_t)t * CC + c];
}

// ---------------- WMMA GEMM body -------------------------------------------
// C = A(f32, MxK) x B (bf16, given transposed as BT[N][K]) (+bias)(+gelu).
// Block tile 128x128, 8 waves (2x4), wave tile 64x32 = 4x2 of 16x16.
// B tile comes in via the Tensor Data Mover; A tile is converted by the VALU.
__device__ __forceinline__ void gemm_body(
    const float* __restrict__ A, const unsigned short* __restrict__ BT,
    const float* __restrict__ bias, float* __restrict__ C,
    int M, int N, int K, int epi)   // epi: 0=none, 1=+bias, 2=+bias+gelu(erf)
{
    __shared__ unsigned short As[128 * 32];   // [m][k] bf16
    __shared__ unsigned short Bs[128 * 32];   // [n][k] bf16 (TDM destination)
    const int tid  = threadIdx.x;
    const int lane = tid & 31, wave = tid >> 5;
    const int wm = wave >> 2, wn = wave & 3;            // 2 x 4 wave grid
    const int bm = blockIdx.y * 128, bn = blockIdx.x * 128;
    const unsigned bs_off = (unsigned)(size_t)(const void*)Bs;  // LDS byte offset

    v8f acc[4][2] = {};

    const int arow = tid >> 1, acol = (tid & 1) * 16;   // 2 threads / row
    const int koff = 8 * (lane >> 4);                   // ISA 16-bit K ordering

    for (int k0 = 0; k0 < K; k0 += 32) {
        __syncthreads();
        // B tile: one TDM descriptor per workgroup, issued by wave 0
        if (wave == 0)
            tdm_load_tile_b(BT + (size_t)bn * K + k0, bs_off, K, N);
        // A tile: 16 fp32 per thread -> bf16 (VALU conversion overlaps the DMA)
        {
            const float* ag = A + (size_t)(bm + arow) * K + k0 + acol;
            if (k0 + 32 < K) __builtin_prefetch(ag + 32, 0, 0);  // next k-slab
            float4 a0 = *(const float4*)(ag + 0);
            float4 a1 = *(const float4*)(ag + 4);
            float4 a2 = *(const float4*)(ag + 8);
            float4 a3 = *(const float4*)(ag + 12);
            uint4 p0 = make_uint4(pk2(a0.x,a0.y), pk2(a0.z,a0.w), pk2(a1.x,a1.y), pk2(a1.z,a1.w));
            uint4 p1 = make_uint4(pk2(a2.x,a2.y), pk2(a2.z,a2.w), pk2(a3.x,a3.y), pk2(a3.z,a3.w));
            *(uint4*)&As[arow * 32 + acol]     = p0;
            *(uint4*)&As[arow * 32 + acol + 8] = p1;
        }
        if (wave == 0) __builtin_amdgcn_s_wait_tensorcnt(0);
        __syncthreads();

        v16bf afrag[4], bfrag[2];
        #pragma unroll
        for (int mi = 0; mi < 4; ++mi) {
            int r = wm * 64 + mi * 16 + (lane & 15);
            v8bf lo = *(const v8bf*)&As[r * 32 + koff];
            v8bf hi = *(const v8bf*)&As[r * 32 + 16 + koff];
            afrag[mi] = __builtin_shufflevector(lo, hi, 0,1,2,3,4,5,6,7,8,9,10,11,12,13,14,15);
        }
        #pragma unroll
        for (int ni = 0; ni < 2; ++ni) {
            int c = wn * 32 + ni * 16 + (lane & 15);
            v8bf lo = *(const v8bf*)&Bs[c * 32 + koff];
            v8bf hi = *(const v8bf*)&Bs[c * 32 + 16 + koff];
            bfrag[ni] = __builtin_shufflevector(lo, hi, 0,1,2,3,4,5,6,7,8,9,10,11,12,13,14,15);
        }
        #pragma unroll
        for (int mi = 0; mi < 4; ++mi)
            #pragma unroll
            for (int ni = 0; ni < 2; ++ni)
                acc[mi][ni] = __builtin_amdgcn_wmma_f32_16x16x32_bf16(
                    false, afrag[mi], false, bfrag[ni], (short)0, acc[mi][ni], false, false);
    }

    // epilogue
    #pragma unroll
    for (int mi = 0; mi < 4; ++mi) {
        int mbase = bm + wm * 64 + mi * 16 + 8 * (lane >> 4);
        #pragma unroll
        for (int ni = 0; ni < 2; ++ni) {
            int col = bn + wn * 32 + ni * 16 + (lane & 15);
            float bv = (epi >= 1) ? bias[col] : 0.0f;
            #pragma unroll
            for (int r = 0; r < 8; ++r) {
                float v = acc[mi][ni][r] + bv;
                if (epi == 2) v = 0.5f * v * (1.0f + erff(v * 0.70710678118f));
                C[(size_t)(mbase + r) * N + col] = v;
            }
        }
    }
}

__global__ __launch_bounds__(256) void gemm_wmma_kernel(
    const float* __restrict__ A, const unsigned short* __restrict__ BT,
    const float* __restrict__ bias, float* __restrict__ C,
    int M, int N, int K, int epi) {
    gemm_body(A, BT, bias, C, M, N, K, epi);
}

// Fused q/k/v projection: gridDim.z = 3 selects weight/bias/output.
// wT slices and out slices are contiguous with the given strides.
__global__ __launch_bounds__(256) void gemm_qkv_kernel(
    const float* __restrict__ A, const unsigned short* __restrict__ wT, size_t wstride,
    const float* __restrict__ bq, const float* __restrict__ bk, const float* __restrict__ bv,
    float* __restrict__ outb, size_t ostride, int M, int N, int K) {
    int z = blockIdx.z;
    const float* bias = (z == 0) ? bq : (z == 1) ? bk : bv;
    gemm_body(A, wT + wstride * z, bias, outb + ostride * z, M, N, K, 1);
}

// ---------------- causal attention (D=16), online softmax ----------------
// grid = (T/256, B*H), block 256; one thread per query row.
__global__ __launch_bounds__(256) void attn_kernel(const float* __restrict__ q,
                                                   const float* __restrict__ k,
                                                   const float* __restrict__ v,
                                                   float* __restrict__ o) {
    __shared__ float Ks[128][16];
    __shared__ float Vs[128][16];
    const int tid = threadIdx.x;
    const int bh = blockIdx.y, b = bh >> 4, h = bh & 15;
    const int tq = blockIdx.x * 256 + tid;
    const size_t qoff = ((size_t)(b * TT + tq)) * CC + h * 16;

    float qv[16];
    #pragma unroll
    for (int d = 0; d < 16; ++d) qv[d] = q[qoff + d];

    float m = -INFINITY, l = 0.0f, acc[16] = {};
    const int lastTile = (blockIdx.x * 256 + 255) >> 7;

    for (int kt = 0; kt <= lastTile; ++kt) {
        int kb = kt * 128;
        __syncthreads();
        {   // stage 128 keys + values (8 floats per thread each)
            int r = tid >> 1, c0 = (tid & 1) * 8;
            size_t off = ((size_t)(b * TT + kb + r)) * CC + h * 16 + c0;
            *(float4*)&Ks[r][c0]     = *(const float4*)(k + off);
            *(float4*)&Ks[r][c0 + 4] = *(const float4*)(k + off + 4);
            *(float4*)&Vs[r][c0]     = *(const float4*)(v + off);
            *(float4*)&Vs[r][c0 + 4] = *(const float4*)(v + off + 4);
        }
        __syncthreads();
        int jmax = tq - kb; if (jmax > 127) jmax = 127;
        for (int j = 0; j <= jmax; ++j) {
            float s = 0.0f;
            #pragma unroll
            for (int d = 0; d < 16; ++d) s += qv[d] * Ks[j][d];
            s *= 0.25f;                         // 1/sqrt(16)
            float nm   = fmaxf(m, s);
            float corr = __expf(m - nm);
            float p    = __expf(s - nm);
            l = l * corr + p;
            #pragma unroll
            for (int d = 0; d < 16; ++d) acc[d] = acc[d] * corr + p * Vs[j][d];
            m = nm;
        }
    }
    float inv = 1.0f / l;
    #pragma unroll
    for (int d = 0; d < 16; ++d) o[qoff + d] = acc[d] * inv;
}

// ---------------- residual add + LayerNorm (wave per row, C=256) -------------
__global__ __launch_bounds__(256) void resln_kernel(float* __restrict__ x,
                                                    const float* __restrict__ res,
                                                    const float* __restrict__ g,
                                                    const float* __restrict__ bta) {
    const int wave = threadIdx.x >> 5, lane = threadIdx.x & 31;
    const int row = blockIdx.x * 8 + wave;
    const size_t base = (size_t)row * CC + lane * 8;
    float vbuf[8], s1 = 0.f, s2 = 0.f;
    #pragma unroll
    for (int i = 0; i < 8; ++i) {
        float t = x[base + i];
        if (res) t += res[base + i];
        vbuf[i] = t; s1 += t; s2 += t * t;
    }
    #pragma unroll
    for (int off = 16; off; off >>= 1) {
        s1 += __shfl_xor(s1, off, 32);
        s2 += __shfl_xor(s2, off, 32);
    }
    float mean = s1 * (1.0f / CC);
    float var  = s2 * (1.0f / CC) - mean * mean;
    float rstd = rsqrtf(var + 1e-5f);
    #pragma unroll
    for (int i = 0; i < 8; ++i) {
        int c = lane * 8 + i;
        x[base + i] = (vbuf[i] - mean) * rstd * g[c] + bta[c];
    }
}

// ---------------- host-side orchestration ----------------
extern "C" void kernel_launch(void* const* d_in, const int* in_sizes, int n_in,
                              void* d_out, int out_size, void* d_ws, size_t ws_size,
                              hipStream_t stream) {
    (void)in_sizes; (void)n_in; (void)out_size; (void)ws_size;
    const int*   idx   = (const int*)  d_in[0];
    const float* tok   = (const float*)d_in[1];
    const float* pos   = (const float*)d_in[2];
    const float* wq    = (const float*)d_in[3];
    const float* bq    = (const float*)d_in[4];
    const float* wk    = (const float*)d_in[5];
    const float* bk    = (const float*)d_in[6];
    const float* wv    = (const float*)d_in[7];
    const float* bv    = (const float*)d_in[8];
    const float* wo    = (const float*)d_in[9];
    const float* bo    = (const float*)d_in[10];
    const float* ln1g  = (const float*)d_in[11];
    const float* ln1b  = (const float*)d_in[12];
    const float* ln2g  = (const float*)d_in[13];
    const float* ln2b  = (const float*)d_in[14];
    const float* w1    = (const float*)d_in[15];
    const float* b1    = (const float*)d_in[16];
    const float* w2    = (const float*)d_in[17];
    const float* b2    = (const float*)d_in[18];
    const float* lnfg  = (const float*)d_in[19];
    const float* lnfb  = (const float*)d_in[20];
    const float* headw = (const float*)d_in[21];
    float* out = (float*)d_out;

    // workspace carve-out (NB: qb/kb/vb and wqT/wkT/wvT must stay contiguous
    // for the fused qkv launch; every size below is a multiple of 256B).
    char* p = (char*)d_ws;
    auto carve = [&](size_t bytes) { void* r = (void*)p; p += (bytes + 255) & ~(size_t)255; return r; };
    float* x    = (float*)carve((size_t)MM * CC * 4);
    float* qb   = (float*)carve((size_t)MM * CC * 4);
    float* kb   = (float*)carve((size_t)MM * CC * 4);
    float* vb   = (float*)carve((size_t)MM * CC * 4);
    float* attb = (float*)carve((size_t)MM * CC * 4);
    float* tmp  = (float*)carve((size_t)MM * CC * 4);
    float* hbuf = (float*)carve((size_t)MM * 4 * CC * 4);
    unsigned short* wqT   = (unsigned short*)carve((size_t)LL * CC * CC * 2);
    unsigned short* wkT   = (unsigned short*)carve((size_t)LL * CC * CC * 2);
    unsigned short* wvT   = (unsigned short*)carve((size_t)LL * CC * CC * 2);
    unsigned short* woT   = (unsigned short*)carve((size_t)LL * CC * CC * 2);
    unsigned short* w1T   = (unsigned short*)carve((size_t)LL * CC * 4 * CC * 2);
    unsigned short* w2T   = (unsigned short*)carve((size_t)LL * 4 * CC * CC * 2);
    unsigned short* headT = (unsigned short*)carve((size_t)VV * CC * 2);

    dim3 blk(256);
    // 1) weight conversion (transpose + bf16)
    convT_kernel<<<dim3(CC * CC / 256, 1, LL), blk, 0, stream>>>(wq, wqT, CC, CC);
    convT_kernel<<<dim3(CC * CC / 256, 1, LL), blk, 0, stream>>>(wk, wkT, CC, CC);
    convT_kernel<<<dim3(CC * CC / 256, 1, LL), blk, 0, stream>>>(wv, wvT, CC, CC);
    convT_kernel<<<dim3(CC * CC / 256, 1, LL), blk, 0, stream>>>(wo, woT, CC, CC);
    convT_kernel<<<dim3(CC * 4 * CC / 256, 1, LL), blk, 0, stream>>>(w1, w1T, CC, 4 * CC);
    convT_kernel<<<dim3(4 * CC * CC / 256, 1, LL), blk, 0, stream>>>(w2, w2T, 4 * CC, CC);
    convT_kernel<<<dim3(CC * VV / 256, 1, 1),  blk, 0, stream>>>(headw, headT, CC, VV);

    // 2) embedding
    embed_kernel<<<dim3(MM * CC / 256), blk, 0, stream>>>(idx, tok, pos, x);

    // 3) transformer layers
    const size_t wstride = (size_t)LL * CC * CC;   // wqT -> wkT -> wvT slice stride
    const size_t ostride = (size_t)MM * CC;        // qb -> kb -> vb stride
    for (int l = 0; l < LL; ++l) {
        gemm_qkv_kernel<<<dim3(CC / 128, MM / 128, 3), blk, 0, stream>>>(
            x, wqT + (size_t)l * CC * CC, wstride, bq + l * CC, bk + l * CC, bv + l * CC,
            qb, ostride, MM, CC, CC);

        attn_kernel<<<dim3(TT / 256, BB * HH), blk, 0, stream>>>(qb, kb, vb, attb);

        gemm_wmma_kernel<<<dim3(CC / 128, MM / 128), blk, 0, stream>>>(
            attb, woT + (size_t)l * CC * CC, bo + l * CC, tmp, MM, CC, CC, 1);
        resln_kernel<<<dim3(MM / 8), blk, 0, stream>>>(x, tmp, ln1g + l * CC, ln1b + l * CC);

        gemm_wmma_kernel<<<dim3(4 * CC / 128, MM / 128), blk, 0, stream>>>(
            x, w1T + (size_t)l * CC * 4 * CC, b1 + l * 4 * CC, hbuf, MM, 4 * CC, CC, 2);
        gemm_wmma_kernel<<<dim3(CC / 128, MM / 128), blk, 0, stream>>>(
            hbuf, w2T + (size_t)l * 4 * CC * CC, b2 + l * CC, tmp, MM, CC, 4 * CC, 1);
        resln_kernel<<<dim3(MM / 8), blk, 0, stream>>>(x, tmp, ln2g + l * CC, ln2b + l * CC);
    }

    // 4) final LN + head GEMM straight into d_out
    resln_kernel<<<dim3(MM / 8), blk, 0, stream>>>(x, nullptr, lnfg, lnfb);
    gemm_wmma_kernel<<<dim3(VV / 128, MM / 128), blk, 0, stream>>>(
        x, headT, nullptr, out, MM, VV, CC, 0);
}
